// graphsage_72069551227476
// MI455X (gfx1250) — matrix operand order repaired and checked by reading, not compiled, not run on hardware
//
#include <hip/hip_runtime.h>
#include <hip/hip_bf16.h>
#include <math.h>

#define NROWS 16384
#define NF 128
#define NCLS 40
#define BM 128
#define BK 32
#define SA 40    // LDS row stride (halfs) for adj tile
#define SB 40    // LDS row stride (halfs) for B tile
#define SS 136   // LDS row stride (halfs) for S / h2 tile
#define SLOG 49  // LDS row stride (floats) for logits

typedef _Float16 f16;
typedef __attribute__((ext_vector_type(16))) _Float16 v16h;
typedef __attribute__((ext_vector_type(8)))  _Float16 v8h;
typedef __attribute__((ext_vector_type(8)))  float    v8f;
typedef __attribute__((ext_vector_type(4)))  float    v4f;
typedef __attribute__((ext_vector_type(4)))  unsigned int v4u;

union F16Frag { v16h v; v8h h[2]; };

static __device__ inline v8f wmma_f16(const F16Frag& a, const F16Frag& b, v8f c) {
  // D = A(16x32 f16) * B(32x16 f16) + C(16x16 f32)
  return __builtin_amdgcn_wmma_f32_16x16x32_f16(false, a.v, false, b.v,
                                                (short)0, c, false, false);
}

// CDNA5 async copy: 32 contiguous bytes global -> LDS, no VGPR staging.
// LDS operand = low 32 bits of generic shared pointer (flat-shared aperture:
// workgroup-relative). INST_OFFSET is added to both
// the global and LDS addresses (ISA 08_async_tensor 4.4), so one address
// pair covers both 16B halves. Tracked by ASYNCcnt.
static __device__ inline void async_copy32(const void* g, void* l) {
  unsigned int       lo = (unsigned int)(unsigned long long)(uintptr_t)l;
  unsigned long long go = (unsigned long long)(uintptr_t)g;
  asm volatile("global_load_async_to_lds_b128 %0, %1, off\n\t"
               "global_load_async_to_lds_b128 %0, %1, off offset:16"
               :: "v"(lo), "v"(go) : "memory");
}
static __device__ inline void wait_async() {
  asm volatile("s_wait_asynccnt 0" ::: "memory");
}

// dst[c*R + r] = (f16)src[r*C + c]
__global__ void cvt_transpose_kernel(const float* __restrict__ src,
                                     f16* __restrict__ dst, int R, int C) {
  int e = blockIdx.x * blockDim.x + threadIdx.x;
  if (e < R * C) {
    int r = e / C, c = e % C;
    dst[(size_t)c * R + r] = (f16)src[e];
  }
}

__global__ __launch_bounds__(256)
void sage_layer_kernel(const float* __restrict__ adj,      // [N][N] f32
                       const float* __restrict__ self_in,  // [N][128] f32 (x or h1)
                       const f16*  __restrict__ featT,     // [128][N] f16 (x^T or h1^T)
                       const f16*  __restrict__ WT,        // [128 out][256 in] f16
                       const float* __restrict__ bias,     // [128]
                       float* __restrict__ h_out,          // [N][128] f32
                       f16*  __restrict__ hT_out,          // [128][N] f16 (or null)
                       int final_stage,
                       const f16*  __restrict__ WlT,       // [40 out][128 in] f16
                       const float* __restrict__ bl,       // [40]
                       float* __restrict__ logp_out,       // [N][40]
                       float* __restrict__ logits_out)     // [N][40]
{
  extern __shared__ char smem[];
  f16*   lds_A = (f16*)(smem);              // 128 x SA halfs (10240 B)
  f16*   lds_B = (f16*)(smem + 10240);      // 128 x SB halfs (10240 B)
  f16*   lds_S = (f16*)(smem + 20480);      // 128 x SS halfs (34816 B)
  float* lds_L = (float*)(smem + 55296);    // 128 x SLOG f32 (25088 B, final only)

  const int tid   = threadIdx.x;
  const int lane  = tid & 31;
  const int wave  = tid >> 5;     // 8 waves; wave owns rows [wave*16, wave*16+16)
  const int l15   = lane & 15;
  const int lhalf = lane >> 4;    // 0/1
  const int m0    = blockIdx.x * BM;

  const int row_st  = tid >> 1;   // cooperative staging: row 0..127
  const int half_st = tid & 1;    // 16-element half of a 32-wide row

  // ---------------- Phase 1: S_raw = (adj * N) @ feat ----------------
  v8f acc[8];
  #pragma unroll
  for (int nt = 0; nt < 8; ++nt)
    #pragma unroll
    for (int v = 0; v < 8; ++v) acc[nt][v] = 0.0f;

  for (int k0 = 0; k0 < NROWS; k0 += BK) {
    __syncthreads();
    // async-stage 32x128 featT tile (pure f16 copy -> ASYNCcnt path, no VGPRs)
    async_copy32(featT + (size_t)row_st * NROWS + k0 + half_st * 16,
                 lds_B + row_st * SB + half_st * 16);
    { // stage 128x32 adj tile, f32 -> f16 scaled by N (avoids f16 denormals)
      const float* srow = adj + (size_t)(m0 + row_st) * NROWS + k0 + half_st * 16;
      const v4f* s = (const v4f*)srow;
      v4f x0 = s[0], x1 = s[1], x2 = s[2], x3 = s[3];
      v8h lo, hi;
      #pragma unroll
      for (int j = 0; j < 4; ++j) {
        lo[j]     = (f16)(x0[j] * 16384.0f);
        lo[4 + j] = (f16)(x1[j] * 16384.0f);
        hi[j]     = (f16)(x2[j] * 16384.0f);
        hi[4 + j] = (f16)(x3[j] * 16384.0f);
      }
      f16* d = lds_A + row_st * SA + half_st * 16;
      *(v8h*)d       = lo;
      *(v8h*)(d + 8) = hi;
      // prefetch next adj chunk (global_prefetch_b8) to hide 64KB-strided rows
      if (k0 + BK < NROWS) __builtin_prefetch(srow + BK, 0, 1);
    }
    wait_async();
    __syncthreads();

    F16Frag a; // A layout: lanes 0-15 K{0..7,16..23}, lanes 16-31 K{8..15,24..31}
    const f16* ap = lds_A + (wave * 16 + l15) * SA + lhalf * 8;
    a.h[0] = *(const v8h*)ap;
    a.h[1] = *(const v8h*)(ap + 16);
    #pragma unroll
    for (int nt = 0; nt < 8; ++nt) {
      F16Frag b; // B layout: lane = n, lanes 0-15 K{0..7,16..23}, lanes 16-31 K{8..15,24..31}
      const f16* bp = lds_B + (nt * 16 + l15) * SB + lhalf * 8;
      b.h[0] = *(const v8h*)bp;
      b.h[1] = *(const v8h*)(bp + 16);
      acc[nt] = wmma_f16(a, b, acc[nt]);
    }
  }

  // ---------------- Phase 2: S = acc / N -> lds_S (layout round-trip) --------
  #pragma unroll
  for (int nt = 0; nt < 8; ++nt)
    #pragma unroll
    for (int v = 0; v < 8; ++v)
      lds_S[(wave * 16 + v + lhalf * 8) * SS + nt * 16 + l15] =
          (f16)(acc[nt][v] * (1.0f / 16384.0f));
  __syncthreads();

  // ---------------- Phase 3: h = relu([self | S] @ W + b) ----------------
  v8f hacc[8];
  #pragma unroll
  for (int nt = 0; nt < 8; ++nt) {
    float bv = bias[nt * 16 + l15];
    #pragma unroll
    for (int v = 0; v < 8; ++v) hacc[nt][v] = bv;
  }
  for (int k2 = 0; k2 < 256; k2 += BK) {
    __syncthreads();
    // async-stage W^T chunk [128 n][32 k]
    async_copy32(WT + (size_t)row_st * 256 + k2 + half_st * 16,
                 lds_B + row_st * SB + half_st * 16);
    wait_async();
    __syncthreads();

    F16Frag a;
    if (k2 < 128) { // self half: convert f32 activations on the fly (L2-hot)
      const float* sp = self_in + (size_t)(m0 + wave * 16 + l15) * NF + k2 + lhalf * 8;
      v4f p0 = *(const v4f*)sp,        p1 = *(const v4f*)(sp + 4);
      v4f p2 = *(const v4f*)(sp + 16), p3 = *(const v4f*)(sp + 20);
      #pragma unroll
      for (int j = 0; j < 4; ++j) {
        a.h[0][j]     = (f16)p0[j];
        a.h[0][4 + j] = (f16)p1[j];
        a.h[1][j]     = (f16)p2[j];
        a.h[1][4 + j] = (f16)p3[j];
      }
    } else {        // aggregated half from lds_S
      const f16* sp = lds_S + (wave * 16 + l15) * SS + (k2 - 128) + lhalf * 8;
      a.h[0] = *(const v8h*)sp;
      a.h[1] = *(const v8h*)(sp + 16);
    }
    #pragma unroll
    for (int nt = 0; nt < 8; ++nt) {
      F16Frag b;
      const f16* bp = lds_B + (nt * 16 + l15) * SB + lhalf * 8;
      b.h[0] = *(const v8h*)bp;
      b.h[1] = *(const v8h*)(bp + 16);
      hacc[nt] = wmma_f16(a, b, hacc[nt]);
    }
  }

  // ---------------- Phase 4: relu + store h (f32 out, f16 transposed ws) ----
  #pragma unroll
  for (int nt = 0; nt < 8; ++nt) {
    #pragma unroll
    for (int v = 0; v < 8; ++v) {
      float hv = fmaxf(hacc[nt][v], 0.0f);
      int ml = wave * 16 + v + lhalf * 8;
      int n  = nt * 16 + l15;
      size_t m = (size_t)(m0 + ml);
      h_out[m * NF + n] = hv;
      if (hT_out) hT_out[(size_t)n * NROWS + m] = (f16)hv;
      if (final_stage) lds_S[ml * SS + n] = (f16)hv; // keep h2 for classifier
    }
  }

  // ---------------- Phase 5/6: logits = h2 @ Wl + bl; log_softmax ----------
  if (final_stage) {
    __syncthreads();
    v8f lacc[3];
    #pragma unroll
    for (int nt = 0; nt < 3; ++nt) {
      int n = nt * 16 + l15;
      float bv = (n < NCLS) ? bl[n] : 0.0f;
      #pragma unroll
      for (int v = 0; v < 8; ++v) lacc[nt][v] = bv;
    }
    for (int k3 = 0; k3 < 128; k3 += BK) {
      __syncthreads();
      if (tid < 96) { // stage Wl^T chunk [48 n (zero-padded)][32 k]
        int n = tid >> 1, half = tid & 1;
        if (n < NCLS) {
          async_copy32(WlT + (size_t)n * NF + k3 + half * 16,
                       lds_B + n * SB + half * 16);
        } else {
          v4u z = {0, 0, 0, 0};
          v4u* d = (v4u*)(lds_B + n * SB + half * 16);
          d[0] = z; d[1] = z;
        }
      }
      wait_async();
      __syncthreads();
      F16Frag a;
      const f16* sp = lds_S + (wave * 16 + l15) * SS + k3 + lhalf * 8;
      a.h[0] = *(const v8h*)sp;
      a.h[1] = *(const v8h*)(sp + 16);
      #pragma unroll
      for (int nt = 0; nt < 3; ++nt) {
        F16Frag b;
        const f16* bp = lds_B + (nt * 16 + l15) * SB + lhalf * 8;
        b.h[0] = *(const v8h*)bp;
        b.h[1] = *(const v8h*)(bp + 16);
        lacc[nt] = wmma_f16(a, b, lacc[nt]);
      }
    }
    #pragma unroll
    for (int nt = 0; nt < 3; ++nt)
      #pragma unroll
      for (int v = 0; v < 8; ++v)
        lds_L[(wave * 16 + v + lhalf * 8) * SLOG + nt * 16 + l15] = lacc[nt][v];
    __syncthreads();
    if (tid < BM) { // one thread per row: log_softmax over 40 classes
      const float* r = lds_L + tid * SLOG;
      float mx = r[0];
      for (int j = 1; j < NCLS; ++j) mx = fmaxf(mx, r[j]);
      float s = 0.0f;
      for (int j = 0; j < NCLS; ++j) s += expf(r[j] - mx);
      float lse = mx + logf(s);
      size_t m = (size_t)(m0 + tid);
      for (int j = 0; j < NCLS; ++j) {
        float lv = r[j];
        logits_out[m * NCLS + j] = lv;
        logp_out[m * NCLS + j]   = lv - lse;
      }
    }
  }
}

extern "C" void kernel_launch(void* const* d_in, const int* in_sizes, int n_in,
                              void* d_out, int out_size, void* d_ws, size_t ws_size,
                              hipStream_t stream) {
  (void)in_sizes; (void)n_in; (void)out_size; (void)ws_size;
  const float* x   = (const float*)d_in[0];
  const float* adj = (const float*)d_in[1];
  const float* W1  = (const float*)d_in[2];
  const float* b1  = (const float*)d_in[3];
  const float* W2  = (const float*)d_in[4];
  const float* b2  = (const float*)d_in[5];
  const float* Wl  = (const float*)d_in[6];
  const float* bl  = (const float*)d_in[7];
  float* out = (float*)d_out;

  char* ws = (char*)d_ws;
  f16* xT  = (f16*)(ws);               // [128][16384] f16, 4 MiB
  f16* h1T = (f16*)(ws + 4194304);     // [128][16384] f16, 4 MiB
  f16* W1T = (f16*)(ws + 8388608);     // [128][256]  f16
  f16* W2T = (f16*)(ws + 8454144);     // [128][256]  f16
  f16* WlT = (f16*)(ws + 8519680);     // [40][128]   f16

  float* logp   = out;                                  // [N][40]
  float* h1     = out + (size_t)NROWS * NCLS;           // [N][128]
  float* h2     = h1  + (size_t)NROWS * NF;             // [N][128]
  float* logits = h2  + (size_t)NROWS * NF;             // [N][40]

  cvt_transpose_kernel<<<(NROWS * NF + 255) / 256, 256, 0, stream>>>(x, xT, NROWS, NF);
  cvt_transpose_kernel<<<(256 * NF + 255) / 256, 256, 0, stream>>>(W1, W1T, 256, NF);
  cvt_transpose_kernel<<<(256 * NF + 255) / 256, 256, 0, stream>>>(W2, W2T, 256, NF);
  cvt_transpose_kernel<<<(NF * NCLS + 255) / 256, 256, 0, stream>>>(Wl, WlT, NF, NCLS);

  // Layer 1: h1 = relu([x | adj@x] @ W1 + b1)
  sage_layer_kernel<<<NROWS / BM, 256, 55296, stream>>>(
      adj, x, xT, W1T, b1, h1, h1T, 0, nullptr, nullptr, nullptr, nullptr);
  // Layer 2 (+classifier + log_softmax)
  sage_layer_kernel<<<NROWS / BM, 256, 80384, stream>>>(
      adj, h1, h1T, W2T, b2, h2, nullptr, 1, WlT, bl, logp, logits);
}
